// BDH_89197880803739
// MI455X (gfx1250) — matrix operand order repaired and checked by reading, not compiled, not run on hardware
//
#include <hip/hip_runtime.h>

// ---------------- model dimensions (from reference) ----------------
#define L_LAYERS 2
#define D_MODEL  256
#define NHEADS   4
#define N_DIM    2048
#define T_SEQ    2048
#define B_BATCH  4
#define VOC_SZ   256
#define N2       (N_DIM/2)
#define GROUPS   (NHEADS/2)
#define EPS_LN   1e-5f
#define SCALE_F  0.022097086912079608f   /* N^-0.5 = 1/sqrt(2048) */
#define TWO_PI_F 6.2831853071795864f

typedef _Float16 half_t;
typedef __attribute__((ext_vector_type(16))) _Float16 v16h;
typedef __attribute__((ext_vector_type(8)))  float    v8f;

#define LDS_K 48   // padded LDS row stride (halves): 96B rows keep 16B alignment, spread banks

// ---------------- CDNA5 async global->LDS copy (ASYNCcnt) ----------------
// GV mode: LDS address comes from the vdst VGPR; global addr = VGPR pair. No VGPR staging.
__device__ __forceinline__ void async_load_b128(unsigned lds_addr, const void* g) {
  asm volatile("global_load_async_to_lds_b128 %0, %1, off"
               :: "v"(lds_addr), "v"((unsigned long long)(uintptr_t)g)
               : "memory");
}
__device__ __forceinline__ void wait_async0() {
  asm volatile("s_wait_asynccnt 0x0" ::: "memory");
}

// ---------------- WMMA fragment loaders (wave32, 16x16x32 f16) ----------------
union FragCast { uint4 u[2]; v16h v; };

// A fragment (16x32): lanes 0-15 = rows M=0..15, K {0..7,16..23};
// lanes 16-31 = rows M=0..15, K {8..15,24..31}.
__device__ __forceinline__ v16h frag_a(const half_t* tile, int row, int kh) {
  FragCast f;
  f.u[0] = *reinterpret_cast<const uint4*>(tile + row * LDS_K + kh * 8);
  f.u[1] = *reinterpret_cast<const uint4*>(tile + row * LDS_K + 16 + kh * 8);
  return f.v;
}
// B fragment (32x16) from LDS tile stored [n][k]: lane = col n (mod 16),
// lanes 0-15: K {0..7,16..23}; lanes 16-31: K {8..15,24..31}.
__device__ __forceinline__ v16h frag_b(const half_t* tile, int col, int kh) {
  FragCast f;
  f.u[0] = *reinterpret_cast<const uint4*>(tile + col * LDS_K + kh * 8);
  f.u[1] = *reinterpret_cast<const uint4*>(tile + col * LDS_K + 16 + kh * 8);
  return f.v;
}

__device__ __forceinline__ float wave_sum(float v) {
#pragma unroll
  for (int o = 16; o > 0; o >>= 1) v += __shfl_xor(v, o, 32);
  return v;
}

// ---------------- generic NT GEMM: C(MxN) = sum_h A_h(MxK) * B_h(NxK)^T ------
// 128x128 block, 4 waves, 64x64 per wave, double-buffered LDS fed by async DMA.
// A rows remapped through (row/seg, row%seg) to handle [b][h][t][*] layouts.
__global__ void __launch_bounds__(128)
gemm_nt(const half_t* __restrict__ A, long lda, long aSeg, long aZ, int aZdiv, long aH,
        const half_t* __restrict__ Bm, long ldb, long bZ, int bZdiv, long bH,
        float* __restrict__ Cf, half_t* __restrict__ Cb, long ldc, long cSeg, long cZ,
        int M, int Nn, int K, int nH, int seg, int relu, int causal)
{
  __shared__ half_t As[2 * 128 * LDS_K];
  __shared__ half_t Bs[2 * 128 * LDS_K];
  const int z = blockIdx.z;
  A  += (long)(z / aZdiv) * aZ;
  Bm += (long)(z / bZdiv) * bZ;
  const long cOffZ = (long)z * cZ;
  const int m0 = blockIdx.y * 128, n0 = blockIdx.x * 128;
  const int Keff = causal ? min(K, m0 + 128) : K;
  const int tid = threadIdx.x;
  const int lane = tid & 31, wave = tid >> 5;
  const int wm = wave >> 1, wn = wave & 1;        // 2x2 wave grid, 64x64 each
  const int lc = lane & 15, kh = lane >> 4;

  // hoisted per-thread addressing (K-invariant): 4 16B chunks per matrix
  int aOff[4], bOff[4];
  unsigned sAdrA[4], sAdrB[4];
#pragma unroll
  for (int it = 0; it < 4; ++it) {
    int c = tid + it * 128;                       // 512 chunks per 128x32 tile
    int row = c >> 2, kg = c & 3;
    long gr = m0 + row;
    aOff[it] = (int)((gr / seg) * aSeg + (gr % seg) * lda + kg * 8);
    bOff[it] = (int)((long)(n0 + row) * ldb + kg * 8);
    int so = row * LDS_K + kg * 8;
    sAdrA[it] = (unsigned)(uintptr_t)(As + so);
    sAdrB[it] = (unsigned)(uintptr_t)(Bs + so);
  }
  const unsigned bufBytes = 128 * LDS_K * sizeof(half_t);

  v8f acc[4][4];
#pragma unroll
  for (int x = 0; x < 4; ++x)
#pragma unroll
    for (int y = 0; y < 4; ++y)
#pragma unroll
      for (int r = 0; r < 8; ++r) acc[x][y][r] = 0.f;

  int hcur = 0, kcur = 0;                         // coords of next tile to fetch
  auto gissue = [&](unsigned dstByteOff) {
    const half_t* Ah = A  + (long)hcur * aH + kcur;
    const half_t* Bh = Bm + (long)hcur * bH + kcur;
#pragma unroll
    for (int it = 0; it < 4; ++it) {
      async_load_b128(sAdrA[it] + dstByteOff, Ah + aOff[it]);
      async_load_b128(sAdrB[it] + dstByteOff, Bh + bOff[it]);
    }
    kcur += 32; if (kcur >= Keff) { kcur = 0; ++hcur; }
  };

  const int total = nH * (Keff >> 5);

  gissue(0);                                      // prologue: tile 0 -> buffer 0
  wait_async0();
  __syncthreads();

  for (int i = 0; i < total; ++i) {
    const int bufH = (i & 1) * (128 * LDS_K);     // element offset for frag reads
    const bool more = (i + 1 < total);
    if (more) gissue(((i + 1) & 1) * bufBytes);   // DMA next tile into other buffer

    const half_t* Ab = As + bufH;
    const half_t* Bb = Bs + bufH;
    v16h af[4];
#pragma unroll
    for (int x = 0; x < 4; ++x)
      af[x] = frag_a(Ab, wm * 64 + x * 16 + lc, kh);
#pragma unroll
    for (int y = 0; y < 4; ++y) {
      v16h bfy = frag_b(Bb, wn * 64 + y * 16 + lc, kh);
#pragma unroll
      for (int x = 0; x < 4; ++x)
        acc[x][y] = __builtin_amdgcn_wmma_f32_16x16x32_f16(
            false, af[x], false, bfy, (short)0, acc[x][y], false, false);
    }

    if (more) {
      wait_async0();
      __syncthreads();
    }
  }

  // C layout: col = n-base + lane%16 ; row = m-base + (lane/16)*8 + r
#pragma unroll
  for (int x = 0; x < 4; ++x) {
#pragma unroll
    for (int y = 0; y < 4; ++y) {
      int gcol = n0 + wn * 64 + y * 16 + lc;
#pragma unroll
      for (int r = 0; r < 8; ++r) {
        long grow = m0 + wm * 64 + x * 16 + kh * 8 + r;
        float v = acc[x][y][r];
        if (relu) v = v > 0.f ? v : 0.f;
        long off = cOffZ + (grow / seg) * cSeg + (grow % seg) * ldc + gcol;
        if (Cb) Cb[off] = (half_t)v;
        else    Cf[off] = v;
      }
    }
  }
}

// ---------------- fused causal score kernel ----------------------------------
// diff = mask * (q1 q1^T - sigmoid(lam) q2 q2^T) * scale ; 64x64 block, dual acc,
// double-buffered LDS fed by async DMA. Blocks with t0+64 <= s0 < t0+128 store
// zeros (read by the causal-clamped outg GEMM); blocks beyond exit immediately.
__global__ void __launch_bounds__(128)
scores_nt(const half_t* __restrict__ QR, const float* __restrict__ lam,
          half_t* __restrict__ diff)
{
  const int t0 = blockIdx.y * 64, s0 = blockIdx.x * 64;
  const int z = blockIdx.z;             // b*GROUPS + g
  half_t* out = diff + (long)z * T_SEQ * T_SEQ;
  const int tid = threadIdx.x;
  if (s0 >= t0 + 128) return;
  if (s0 >= t0 + 64) {                  // zero band read by outg's K clamp
    uint4 zz = {0, 0, 0, 0};
#pragma unroll
    for (int it = 0; it < 4; ++it) {
      int c = tid + it * 128;           // 512 chunks of 16B = 64x64 halves
      int row = c >> 3, kg = c & 7;
      *reinterpret_cast<uint4*>(out + (long)(t0 + row) * T_SEQ + s0 + kg * 8) = zz;
    }
    return;
  }
  __shared__ half_t A1s[2 * 64 * LDS_K], A2s[2 * 64 * LDS_K];
  __shared__ half_t B1s[2 * 64 * LDS_K], B2s[2 * 64 * LDS_K];
  const int b = z / GROUPS, g = z % GROUPS;
  const half_t* q1 = QR + (long)(b * NHEADS + 2 * g) * T_SEQ * N_DIM;
  const half_t* q2 = q1 + (long)T_SEQ * N_DIM;
  const int lane = tid & 31, wave = tid >> 5;
  const int wm = wave >> 1, wn = wave & 1;
  const int lc = lane & 15, kh = lane >> 4;

  int aOff[2], bOff[2];
  unsigned sA1[2], sA2[2], sB1[2], sB2[2];
#pragma unroll
  for (int it = 0; it < 2; ++it) {
    int c = tid + it * 128;             // 256 chunks per 64x32 tile
    int row = c >> 2, kg = c & 3;
    aOff[it] = (t0 + row) * N_DIM + kg * 8;
    bOff[it] = (s0 + row) * N_DIM + kg * 8;
    int so = row * LDS_K + kg * 8;
    sA1[it] = (unsigned)(uintptr_t)(A1s + so);
    sA2[it] = (unsigned)(uintptr_t)(A2s + so);
    sB1[it] = (unsigned)(uintptr_t)(B1s + so);
    sB2[it] = (unsigned)(uintptr_t)(B2s + so);
  }
  const unsigned bufBytes = 64 * LDS_K * sizeof(half_t);

  v8f c1[2][2], c2[2][2];
#pragma unroll
  for (int i = 0; i < 2; ++i)
#pragma unroll
    for (int j = 0; j < 2; ++j)
#pragma unroll
      for (int r = 0; r < 8; ++r) { c1[i][j][r] = 0.f; c2[i][j][r] = 0.f; }

  int kcur = 0;
  auto gissue = [&](unsigned dstByteOff) {
#pragma unroll
    for (int it = 0; it < 2; ++it) {
      async_load_b128(sA1[it] + dstByteOff, q1 + aOff[it] + kcur);
      async_load_b128(sA2[it] + dstByteOff, q2 + aOff[it] + kcur);
      async_load_b128(sB1[it] + dstByteOff, q1 + bOff[it] + kcur);
      async_load_b128(sB2[it] + dstByteOff, q2 + bOff[it] + kcur);
    }
    kcur += 32;
  };

  const int total = N_DIM >> 5;
  gissue(0);
  wait_async0();
  __syncthreads();

  for (int i = 0; i < total; ++i) {
    const int bufH = (i & 1) * (64 * LDS_K);
    const bool more = (i + 1 < total);
    if (more) gissue(((i + 1) & 1) * bufBytes);

    v16h a1f[2], a2f[2];
#pragma unroll
    for (int x = 0; x < 2; ++x) {
      a1f[x] = frag_a(A1s + bufH, wm * 32 + x * 16 + lc, kh);
      a2f[x] = frag_a(A2s + bufH, wm * 32 + x * 16 + lc, kh);
    }
#pragma unroll
    for (int y = 0; y < 2; ++y) {
      v16h b1y = frag_b(B1s + bufH, wn * 32 + y * 16 + lc, kh);
      v16h b2y = frag_b(B2s + bufH, wn * 32 + y * 16 + lc, kh);
#pragma unroll
      for (int x = 0; x < 2; ++x) {
        c1[x][y] = __builtin_amdgcn_wmma_f32_16x16x32_f16(false, a1f[x], false, b1y, (short)0, c1[x][y], false, false);
        c2[x][y] = __builtin_amdgcn_wmma_f32_16x16x32_f16(false, a2f[x], false, b2y, (short)0, c2[x][y], false, false);
      }
    }

    if (more) {
      wait_async0();
      __syncthreads();
    }
  }

  const float lam_s = 1.f / (1.f + expf(-lam[g]));
#pragma unroll
  for (int x = 0; x < 2; ++x)
#pragma unroll
    for (int y = 0; y < 2; ++y) {
      int gs = s0 + wn * 32 + y * 16 + lc;
#pragma unroll
      for (int r = 0; r < 8; ++r) {
        int gt = t0 + wm * 32 + x * 16 + kh * 8 + r;
        float v = (gs < gt) ? (c1[x][y][r] - lam_s * c2[x][y][r]) * SCALE_F : 0.f;
        out[(long)gt * T_SEQ + gs] = (half_t)v;
      }
    }
}

// ---------------- small/elementwise kernels ----------------
__global__ void __launch_bounds__(256)
cvt_transpose(const float* __restrict__ in, half_t* __restrict__ out, int R, int C)
{
  long base = (long)blockIdx.z * R * C;
  int idx = blockIdx.x * 256 + threadIdx.x;
  if (idx >= R * C) return;
  int r = idx / C, c = idx % C;
  out[base + (long)c * R + r] = (half_t)in[base + idx];
}

__global__ void __launch_bounds__(256)
rope_tables(float* __restrict__ cosT, float* __restrict__ sinT)
{
  int idx = blockIdx.x * 256 + threadIdx.x;
  if (idx >= T_SEQ * N2) return;
  int i = idx % N2, t = idx / N2;
  float freq = exp2f(-16.0f * (2.0f * i) / (float)N_DIM) / TWO_PI_F;
  float ang = fmodf((float)t * freq, 1.0f) * TWO_PI_F;
  cosT[idx] = cosf(ang);
  sinT[idx] = sinf(ang);
}

__global__ void __launch_bounds__(256)
embed_ln(const int* __restrict__ idx, const float* __restrict__ emb,
         float* __restrict__ xf, half_t* __restrict__ xh)
{
  int wave = threadIdx.x >> 5, lane = threadIdx.x & 31;
  long row = (long)blockIdx.x * 8 + wave;          // B*T rows
  const float* e = emb + (long)idx[row] * D_MODEL;
  float v[8]; float s = 0.f;
#pragma unroll
  for (int j = 0; j < 8; ++j) { v[j] = e[lane + 32 * j]; s += v[j]; }
  float m = wave_sum(s) * (1.f / D_MODEL);
  float vs = 0.f;
#pragma unroll
  for (int j = 0; j < 8; ++j) { float d = v[j] - m; vs += d * d; }
  float inv = rsqrtf(wave_sum(vs) * (1.f / D_MODEL) + EPS_LN);
#pragma unroll
  for (int j = 0; j < 8; ++j) {
    float y = (v[j] - m) * inv;
    long o = row * D_MODEL + lane + 32 * j;
    xf[o] = y; xh[o] = (half_t)y;
  }
}

__global__ void __launch_bounds__(256)
transpose_x(const half_t* __restrict__ xh, half_t* __restrict__ xT)
{
  long i = (long)blockIdx.x * 256 + threadIdx.x;   // B*T*D
  int d = (int)(i % D_MODEL);
  long bt = i / D_MODEL;
  int t = (int)(bt % T_SEQ);
  long b = bt / T_SEQ;
  xT[(b * D_MODEL + d) * T_SEQ + t] = xh[i];
}

// 4 RoPE pairs (8 halves) per thread; cos/sin via float4
__global__ void __launch_bounds__(256)
rope_apply(const half_t* __restrict__ xs, const float* __restrict__ cosT,
           const float* __restrict__ sinT, half_t* __restrict__ qr)
{
  long p0 = ((long)blockIdx.x * 256 + threadIdx.x) * 4;  // base pair index
  int ip = (int)(p0 % N2);
  int t = (int)((p0 / N2) % T_SEQ);
  float4 c = *reinterpret_cast<const float4*>(cosT + (long)t * N2 + ip);
  float4 s = *reinterpret_cast<const float4*>(sinT + (long)t * N2 + ip);
  uint4 xv = *reinterpret_cast<const uint4*>(xs + 2 * p0);
  const half_t* xp = reinterpret_cast<const half_t*>(&xv);
  uint4 ov;
  half_t* op = reinterpret_cast<half_t*>(&ov);
  float cc[4] = {c.x, c.y, c.z, c.w}, ss[4] = {s.x, s.y, s.z, s.w};
#pragma unroll
  for (int j = 0; j < 4; ++j) {
    float xr = (float)xp[2 * j], xi = (float)xp[2 * j + 1];
    op[2 * j]     = (half_t)(xr * cc[j] - xi * ss[j]);
    op[2 * j + 1] = (half_t)(xr * ss[j] + xi * cc[j]);
  }
  *reinterpret_cast<uint4*>(qr + 2 * p0) = ov;
}

__global__ void __launch_bounds__(256)
ln_outg(const float* __restrict__ outg, half_t* __restrict__ yKV)
{
  int wave = threadIdx.x >> 5, lane = threadIdx.x & 31;
  long row = (long)blockIdx.x * 8 + wave;          // B*GROUPS*T rows
  const float* ir = outg + row * D_MODEL;
  float v[8]; float s = 0.f;
#pragma unroll
  for (int j = 0; j < 8; ++j) { v[j] = ir[lane + 32 * j]; s += v[j]; }
  float m = wave_sum(s) * (1.f / D_MODEL);
  float vs = 0.f;
#pragma unroll
  for (int j = 0; j < 8; ++j) { float d = v[j] - m; vs += d * d; }
  float inv = rsqrtf(wave_sum(vs) * (1.f / D_MODEL) + EPS_LN);
  int t = (int)(row % T_SEQ);
  long bg = row / T_SEQ;
  int g = (int)(bg % GROUPS);
  long b = bg / GROUPS;
  long h0 = ((b * NHEADS + 2 * g) * (long)T_SEQ + t) * D_MODEL;
#pragma unroll
  for (int j = 0; j < 8; ++j) {
    half_t hv = (half_t)((v[j] - m) * inv);
    yKV[h0 + lane + 32 * j] = hv;                               // head 2g
    yKV[h0 + (long)T_SEQ * D_MODEL + lane + 32 * j] = hv;       // head 2g+1
  }
}

// 8 halves per thread
__global__ void __launch_bounds__(256)
xy_mult(const half_t* __restrict__ xs, half_t* __restrict__ ys)
{
  long i0 = ((long)blockIdx.x * 256 + threadIdx.x) * 8;
  uint4 av = *reinterpret_cast<const uint4*>(xs + i0);
  uint4 bv = *reinterpret_cast<uint4*>(ys + i0);
  const half_t* ap = reinterpret_cast<const half_t*>(&av);
  half_t* bp = reinterpret_cast<half_t*>(&bv);
#pragma unroll
  for (int j = 0; j < 8; ++j) bp[j] = (half_t)((float)ap[j] * (float)bp[j]);
  *reinterpret_cast<uint4*>(ys + i0) = bv;
}

__global__ void __launch_bounds__(256)
residual_ln(const float* __restrict__ yMLP, float* __restrict__ xf, half_t* __restrict__ xh)
{
  int wave = threadIdx.x >> 5, lane = threadIdx.x & 31;
  long row = (long)blockIdx.x * 8 + wave;          // B*T rows
  const float* yr = yMLP + row * D_MODEL;
  float v[8]; float s = 0.f;
#pragma unroll
  for (int j = 0; j < 8; ++j) { v[j] = yr[lane + 32 * j]; s += v[j]; }
  float m = wave_sum(s) * (1.f / D_MODEL);
  float vs = 0.f;
#pragma unroll
  for (int j = 0; j < 8; ++j) { float d = v[j] - m; vs += d * d; }
  float inv = rsqrtf(wave_sum(vs) * (1.f / D_MODEL) + EPS_LN);
  float u[8]; float s2 = 0.f;
#pragma unroll
  for (int j = 0; j < 8; ++j) {
    u[j] = xf[row * D_MODEL + lane + 32 * j] + (v[j] - m) * inv;
    s2 += u[j];
  }
  float m2 = wave_sum(s2) * (1.f / D_MODEL);
  float vs2 = 0.f;
#pragma unroll
  for (int j = 0; j < 8; ++j) { float d = u[j] - m2; vs2 += d * d; }
  float inv2 = rsqrtf(wave_sum(vs2) * (1.f / D_MODEL) + EPS_LN);
#pragma unroll
  for (int j = 0; j < 8; ++j) {
    float y = (u[j] - m2) * inv2;
    long o = row * D_MODEL + lane + 32 * j;
    xf[o] = y; xh[o] = (half_t)y;
  }
}

// ---------------- host orchestration ----------------
extern "C" void kernel_launch(void* const* d_in, const int* in_sizes, int n_in,
                              void* d_out, int out_size, void* d_ws, size_t ws_size,
                              hipStream_t stream)
{
  (void)in_sizes; (void)n_in; (void)out_size; (void)ws_size;
  const int*   idx  = (const int*)d_in[0];
  const float* emb  = (const float*)d_in[1];
  const float* lmh  = (const float*)d_in[2];
  const float* enc  = (const float*)d_in[3];
  const float* encv = (const float*)d_in[4];
  const float* dec  = (const float*)d_in[5];
  const float* lam  = (const float*)d_in[6];

  const long TN = (long)T_SEQ * N_DIM;
  const long TD = (long)T_SEQ * D_MODEL;
  const long ND = (long)N_DIM * D_MODEL;
  const long TT = (long)T_SEQ * T_SEQ;

  char* w = (char*)d_ws;
  auto alloc = [&](size_t bytes) -> char* {
    char* p = w; w += (bytes + 255) & ~(size_t)255; return p;
  };
  half_t* encT  = (half_t*)alloc(sizeof(half_t) * L_LAYERS * NHEADS * ND);
  half_t* encvT = (half_t*)alloc(sizeof(half_t) * L_LAYERS * NHEADS * ND);
  half_t* decT  = (half_t*)alloc(sizeof(half_t) * L_LAYERS * NHEADS * ND);
  half_t* lmhT  = (half_t*)alloc(sizeof(half_t) * D_MODEL * VOC_SZ);
  float*  cosT  = (float*)alloc(sizeof(float) * (long)T_SEQ * N2);
  float*  sinT  = (float*)alloc(sizeof(float) * (long)T_SEQ * N2);
  float*  xf    = (float*)alloc(sizeof(float) * B_BATCH * TD);
  half_t* xh    = (half_t*)alloc(sizeof(half_t) * B_BATCH * TD);
  half_t* xT    = (half_t*)alloc(sizeof(half_t) * B_BATCH * TD);
  half_t* xs    = (half_t*)alloc(sizeof(half_t) * B_BATCH * NHEADS * TN);
  half_t* qr    = (half_t*)alloc(sizeof(half_t) * B_BATCH * NHEADS * TN); // reused as ys/xy
  half_t* dif   = (half_t*)alloc(sizeof(half_t) * B_BATCH * GROUPS * TT);
  float*  outg  = (float*)alloc(sizeof(float) * B_BATCH * GROUPS * TD);
  half_t* yKV   = (half_t*)alloc(sizeof(half_t) * B_BATCH * NHEADS * TD);
  float*  yMLP  = (float*)alloc(sizeof(float) * B_BATCH * TD);

  // ---- weight prep / tables / embedding ----
  cvt_transpose<<<dim3((D_MODEL * N_DIM + 255) / 256, 1, L_LAYERS * NHEADS), 256, 0, stream>>>(enc,  encT,  D_MODEL, N_DIM);
  cvt_transpose<<<dim3((D_MODEL * N_DIM + 255) / 256, 1, L_LAYERS * NHEADS), 256, 0, stream>>>(encv, encvT, D_MODEL, N_DIM);
  cvt_transpose<<<dim3((N_DIM * D_MODEL + 255) / 256, 1, L_LAYERS * NHEADS), 256, 0, stream>>>(dec,  decT,  N_DIM,  D_MODEL);
  cvt_transpose<<<dim3((D_MODEL * VOC_SZ + 255) / 256, 1, 1), 256, 0, stream>>>(lmh, lmhT, D_MODEL, VOC_SZ);
  rope_tables<<<(T_SEQ * N2 + 255) / 256, 256, 0, stream>>>(cosT, sinT);
  embed_ln<<<(B_BATCH * T_SEQ) / 8, 256, 0, stream>>>(idx, emb, xf, xh);

  for (int l = 0; l < L_LAYERS; ++l) {
    half_t* encTl  = encT  + (long)l * NHEADS * ND;
    half_t* encvTl = encvT + (long)l * NHEADS * ND;
    half_t* decTl  = decT  + (long)l * NHEADS * ND;

    transpose_x<<<(int)(B_BATCH * TD / 256), 256, 0, stream>>>(xh, xT);

    // xs = relu(x @ enc[h])            [z = head]
    gemm_nt<<<dim3(N_DIM / 128, (B_BATCH * T_SEQ) / 128, NHEADS), 128, 0, stream>>>(
        xh, D_MODEL, TD, 0, 1, 0,
        encTl, D_MODEL, ND, 1, 0,
        nullptr, xs, N_DIM, (long)NHEADS * TN, TN,
        B_BATCH * T_SEQ, N_DIM, D_MODEL, 1, T_SEQ, 1, 0);

    rope_apply<<<(int)((long)B_BATCH * NHEADS * T_SEQ * N2 / 4 / 256), 256, 0, stream>>>(xs, cosT, sinT, qr);

    // diff = mask * (q1 q1^T - sigmoid(lam) q2 q2^T) * scale   [z = b*G+g]
    scores_nt<<<dim3(T_SEQ / 64, T_SEQ / 64, B_BATCH * GROUPS), 128, 0, stream>>>(qr, lam, dif);

    // outg = diff @ x   (B operand = xT, causal K clamp)       [z = b*G+g]
    gemm_nt<<<dim3(D_MODEL / 128, T_SEQ / 128, B_BATCH * GROUPS), 128, 0, stream>>>(
        dif, T_SEQ, 0, TT, 1, 0,
        xT, T_SEQ, TD, GROUPS, 0,
        outg, nullptr, D_MODEL, 0, TD,
        T_SEQ, D_MODEL, T_SEQ, 1, T_SEQ, 0, 1);

    ln_outg<<<(B_BATCH * GROUPS * T_SEQ) / 8, 256, 0, stream>>>(outg, yKV);

    // ys = relu(yKV @ encv[h]) -> qr buffer                    [z = head]
    gemm_nt<<<dim3(N_DIM / 128, (B_BATCH * T_SEQ) / 128, NHEADS), 128, 0, stream>>>(
        yKV, D_MODEL, (long)NHEADS * TD, TD, 1, 0,
        encvTl, D_MODEL, ND, 1, 0,
        nullptr, qr, N_DIM, (long)NHEADS * TN, TN,
        B_BATCH * T_SEQ, N_DIM, D_MODEL, 1, T_SEQ, 1, 0);

    // xy = xs * ys (in place in qr)
    xy_mult<<<(int)((long)B_BATCH * NHEADS * TN / 8 / 256), 256, 0, stream>>>(xs, qr);

    // yMLP = sum_h xy[h] @ dec[h]
    gemm_nt<<<dim3(D_MODEL / 128, (B_BATCH * T_SEQ) / 128, 1), 128, 0, stream>>>(
        qr, N_DIM, (long)NHEADS * TN, 0, 1, TN,
        decTl, N_DIM, 0, 1, ND,
        yMLP, nullptr, D_MODEL, TD, 0,
        B_BATCH * T_SEQ, D_MODEL, N_DIM, NHEADS, T_SEQ, 0, 0);

    residual_ln<<<(B_BATCH * T_SEQ) / 8, 256, 0, stream>>>(yMLP, xf, xh);
  }

  // out = x @ lm_head  (f32 output)
  gemm_nt<<<dim3(VOC_SZ / 128, (B_BATCH * T_SEQ) / 128, 1), 128, 0, stream>>>(
      xh, D_MODEL, TD, 0, 1, 0,
      lmhT, D_MODEL, 0, 1, 0,
      (float*)d_out, nullptr, VOC_SZ, (long)T_SEQ * VOC_SZ, 0,
      B_BATCH * T_SEQ, VOC_SZ, D_MODEL, 1, T_SEQ, 0, 0);
}
